// BDA_ENC_85160611545360
// MI455X (gfx1250) — compile-verified, matrix operand order checked by reading, not run on hardware
//
#include <hip/hip_runtime.h>
#include <hip/hip_bf16.h>
#include <math.h>

// ---------------------------------------------------------------------------
// BEV deformable-DETR decoder forward for MI455X (gfx1250, wave32, WMMA).
//
// Roofline reasoning (MI455X: 23.3 TB/s HBM, 192MB L2, wave32 WGP, WMMA):
//  * GEMMs: ~45 GFLOP/fwd, all M=8192 -> bf16 WMMA f32-acc, LDS-tiled 128x64,
//    double-buffered (1 barrier/K-step, global fetch overlaps WMMA).
//  * Attention: flash-style per (b,h); scores never hit HBM (saves 768MB/layer).
//  * Deform: vproj fused into bilinear sampling (saves 654MB/layer HBM traffic).
//  * All LDS staging packed as bf16x2 dwords (ds_store_b32, global_load_b64).
// ---------------------------------------------------------------------------

#define D       256
#define NQ      1024
#define BATCH   8
#define NH      8
#define HD      32
#define PP      8
#define HBEV    200
#define WBEV    200
#define MROWS   (BATCH * NQ)          /* 8192 */

typedef __attribute__((ext_vector_type(16))) __bf16 v16bf;
typedef __attribute__((ext_vector_type(8)))  float  v8f;

union BF16x16 {
  v16bf v;
  unsigned short s[16];
  unsigned int   u[8];
};

struct __attribute__((aligned(16))) U4 { unsigned int x, y, z, w; };
struct __attribute__((aligned(8)))  F2 { float x, y; };

__device__ __forceinline__ unsigned short f2bf(float f) {
  union { float f; unsigned int u; } c; c.f = f;
  unsigned int u = c.u;
  u += 0x7FFFu + ((u >> 16) & 1u);          // round-to-nearest-even
  return (unsigned short)(u >> 16);
}
__device__ __forceinline__ unsigned int pack2(float lo, float hi) {
  return (unsigned int)f2bf(lo) | ((unsigned int)f2bf(hi) << 16);
}

// Load a 16x32(bf16) A/B WMMA fragment from LDS.
// Lane layout (CDNA5 ISA 7.12.2): lane<16 holds K=0..7,16..23 ; lane>=16
// holds K=8..15,24..31 -> two ds_load_b128 per lane from a row-major LDS row.
__device__ __forceinline__ void load_frag(BF16x16& fr, const unsigned short* p) {
  U4 a = *(const U4*)(p);        // K base .. base+7
  U4 b = *(const U4*)(p + 16);   // K base+16 .. base+23
  fr.u[0] = a.x; fr.u[1] = a.y; fr.u[2] = a.z; fr.u[3] = a.w;
  fr.u[4] = b.x; fr.u[5] = b.y; fr.u[6] = b.z; fr.u[7] = b.w;
}

#define WMMA_BF16(A, B, C) \
  __builtin_amdgcn_wmma_f32_16x16x32_bf16(false, (A), false, (B), (short)0, (C), false, false)

// ---------------------------------------------------------------------------
// Generic GEMM: C[M,N] = act(A[M,K] @ W[K,N] + bias), bf16 WMMA, f32 acc.
// Block: 256 thr (8 waves). Tile 128(M) x 64(N), K-step 32, double-buffered.
// Wave w owns rows 16w..16w+15 and all 64 N columns (4 WMMA tiles).
// ---------------------------------------------------------------------------
__global__ __launch_bounds__(256) void gemm_bf16(
    const float* __restrict__ A, const float* __restrict__ W,
    const float* __restrict__ bias, float* __restrict__ C,
    int M, int N, int K, int relu)
{
  __shared__ __attribute__((aligned(16))) unsigned short As[2][128][40]; // padded
  __shared__ __attribute__((aligned(16))) unsigned short Bs[2][64][40];  // W^T tile

  const int tid   = threadIdx.x;
  const int wave  = tid >> 5;
  const int lane  = tid & 31;
  const int lhalf = lane >> 4;
  const int lm    = lane & 15;
  const int m0 = blockIdx.y * 128;
  const int n0 = blockIdx.x * 64;

  v8f acc[4] = {};
  unsigned int aReg[8];   // packed bf16x2: A tile slice held by this thread
  unsigned int bReg[4];   // packed bf16x2: W tile slice (transposed pairs)

  // Fetch one 128x32 A tile + 32x64 W tile into registers (packed bf16x2).
  auto fetch = [&](int k0) {
    #pragma unroll
    for (int i = 0; i < 8; ++i) {
      int li = i * 256 + tid;
      int r = li >> 4, c2 = li & 15;                  // row, pair-column
      F2 f = *(const F2*)&A[(size_t)(m0 + r) * K + k0 + 2 * c2];
      aReg[i] = pack2(f.x, f.y);
    }
    #pragma unroll
    for (int i = 0; i < 4; ++i) {
      int li = i * 256 + tid;
      int n = li & 63, kp = li >> 6;                  // n-col, k-pair
      float w0 = W[(size_t)(k0 + 2 * kp)     * N + n0 + n];
      float w1 = W[(size_t)(k0 + 2 * kp + 1) * N + n0 + n];
      bReg[i] = pack2(w0, w1);
    }
  };
  // Stage registers into LDS buffer `buf` as dword stores.
  auto stage = [&](int buf) {
    #pragma unroll
    for (int i = 0; i < 8; ++i) {
      int li = i * 256 + tid;
      int r = li >> 4, c2 = li & 15;
      *(unsigned int*)&As[buf][r][2 * c2] = aReg[i];
    }
    #pragma unroll
    for (int i = 0; i < 4; ++i) {
      int li = i * 256 + tid;
      int n = li & 63, kp = li >> 6;
      *(unsigned int*)&Bs[buf][n][2 * kp] = bReg[i];  // W^T: Bs[n][k]
    }
  };

  fetch(0); stage(0); __syncthreads();
  int cb = 0;
  for (int k0 = 0;; k0 += 32) {
    const bool more = (k0 + 32 < K);
    if (more) fetch(k0 + 32);                         // overlap with WMMA below

    BF16x16 af;
    load_frag(af, &As[cb][wave * 16 + lm][lhalf * 8]);
    #pragma unroll
    for (int j = 0; j < 4; ++j) {
      BF16x16 bfrag;
      load_frag(bfrag, &Bs[cb][j * 16 + lm][lhalf * 8]);
      acc[j] = WMMA_BF16(af.v, bfrag.v, acc[j]);
    }
    if (!more) break;
    stage(cb ^ 1);
    __syncthreads();                                  // single barrier per step
    cb ^= 1;
  }

  // Epilogue: D layout -> row = v + 8*lhalf, col = lm (per 16x16 tile).
  #pragma unroll
  for (int j = 0; j < 4; ++j) {
    #pragma unroll
    for (int v = 0; v < 8; ++v) {
      int row = m0 + wave * 16 + v + 8 * lhalf;
      int col = n0 + j * 16 + lm;
      float val = acc[j][v] + bias[col];
      if (relu) val = fmaxf(val, 0.f);
      C[(size_t)row * N + col] = val;
    }
  }
}

// ---------------------------------------------------------------------------
// Flash attention, one (b,h) per (blockIdx.z, blockIdx.y), 128 queries/block,
// 8 waves x 16 queries. S^T = K·Q^T (one WMMA per 16key x 16q, contraction=HD),
// online softmax over key chunks of 32, O^T += V^T · P. The exp'd S^T
// D-fragment maps element-for-element onto the P B-fragment (no shuffles).
// K/V staging is double-buffered, packed bf16x2.
// ---------------------------------------------------------------------------
__global__ __launch_bounds__(256) void attn_flash(
    const float* __restrict__ Q, const float* __restrict__ K,
    const float* __restrict__ V, float* __restrict__ O)
{
  __shared__ __attribute__((aligned(16))) unsigned short Ks[2][32][40];   // key x hd
  __shared__ __attribute__((aligned(16))) unsigned short VTs[2][32][40];  // hd x key

  const int b = blockIdx.z, h = blockIdx.y, qg = blockIdx.x;
  const int tid = threadIdx.x;
  const int wave = tid >> 5, lane = tid & 31;
  const int lhalf = lane >> 4, lm = lane & 15;
  const int q_ = qg * 128 + wave * 16 + lm;
  const float qscale = 0.17677669529663687f;   // 1/sqrt(HD)

  // Q^T B-fragment: column = query (lm), K index = hd (CDNA5 mapping).
  BF16x16 qf;
  const float* qrow = Q + ((size_t)(b * NQ + q_) * D + h * HD);
  #pragma unroll
  for (int e = 0; e < 16; ++e) {
    int hd = 8 * lhalf + (e < 8 ? e : e + 8);
    qf.s[e] = f2bf(qrow[hd] * qscale);
  }

  unsigned int kReg[2], vReg[2];
  auto fetch = [&](int kc) {
    #pragma unroll
    for (int i = 0; i < 2; ++i) {                       // K: 32 keys x 16 hd-pairs
      int li = i * 256 + tid;
      int key = li >> 4, hp = li & 15;
      F2 f = *(const F2*)&K[(size_t)(b * NQ + kc + key) * D + h * HD + 2 * hp];
      kReg[i] = pack2(f.x, f.y);
    }
    #pragma unroll
    for (int i = 0; i < 2; ++i) {                       // V^T: 32 hd x 16 key-pairs
      int li = i * 256 + tid;
      int hd = li & 31, kp = li >> 5;
      float v0 = V[(size_t)(b * NQ + kc + 2 * kp)     * D + h * HD + hd];
      float v1 = V[(size_t)(b * NQ + kc + 2 * kp + 1) * D + h * HD + hd];
      vReg[i] = pack2(v0, v1);
    }
  };
  auto stage = [&](int buf) {
    #pragma unroll
    for (int i = 0; i < 2; ++i) {
      int li = i * 256 + tid;
      int key = li >> 4, hp = li & 15;
      *(unsigned int*)&Ks[buf][key][2 * hp] = kReg[i];
    }
    #pragma unroll
    for (int i = 0; i < 2; ++i) {
      int li = i * 256 + tid;
      int hd = li & 31, kp = li >> 5;
      *(unsigned int*)&VTs[buf][hd][2 * kp] = vReg[i];
    }
  };

  v8f o0 = {}, o1 = {};
  float m_run = -1e30f, l_run = 0.f;

  fetch(0); stage(0); __syncthreads();
  int cb = 0;
  for (int kc = 0;; kc += 32) {
    const bool more = (kc + 32 < NQ);
    if (more) fetch(kc + 32);                           // overlap with compute

    BF16x16 a0, a1;
    load_frag(a0, &Ks[cb][lm][lhalf * 8]);
    load_frag(a1, &Ks[cb][16 + lm][lhalf * 8]);
    v8f s0 = {}, s1 = {};
    s0 = WMMA_BF16(a0.v, qf.v, s0);   // keys 0..15  x queries
    s1 = WMMA_BF16(a1.v, qf.v, s1);   // keys 16..31 x queries

    // Online softmax per query column (lanes lm and lm+16 share a query).
    float mloc = s0[0];
    #pragma unroll
    for (int v = 1; v < 8; ++v) mloc = fmaxf(mloc, s0[v]);
    #pragma unroll
    for (int v = 0; v < 8; ++v) mloc = fmaxf(mloc, s1[v]);
    mloc = fmaxf(mloc, __shfl_xor(mloc, 16, 32));
    float mnew  = fmaxf(m_run, mloc);
    float scale = __expf(m_run - mnew);

    float p0[8], p1[8], ssum = 0.f;
    #pragma unroll
    for (int v = 0; v < 8; ++v) { p0[v] = __expf(s0[v] - mnew); ssum += p0[v]; }
    #pragma unroll
    for (int v = 0; v < 8; ++v) { p1[v] = __expf(s1[v] - mnew); ssum += p1[v]; }
    ssum += __shfl_xor(ssum, 16, 32);
    l_run = l_run * scale + ssum;
    m_run = mnew;
    #pragma unroll
    for (int v = 0; v < 8; ++v) { o0[v] *= scale; o1[v] *= scale; }

    // P B-fragment: e<8 -> keys 0..15 half (=p0[e]); e>=8 -> keys 16..31 (=p1).
    BF16x16 pf;
    #pragma unroll
    for (int e = 0; e < 16; ++e) pf.s[e] = f2bf(e < 8 ? p0[e] : p1[e - 8]);

    BF16x16 va0, va1;
    load_frag(va0, &VTs[cb][lm][lhalf * 8]);
    load_frag(va1, &VTs[cb][16 + lm][lhalf * 8]);
    o0 = WMMA_BF16(va0.v, pf.v, o0);  // hd 0..15  x queries
    o1 = WMMA_BF16(va1.v, pf.v, o1);  // hd 16..31 x queries

    if (!more) break;
    stage(cb ^ 1);
    __syncthreads();
    cb ^= 1;
  }

  float inv = 1.f / l_run;
  #pragma unroll
  for (int v = 0; v < 8; ++v) {
    int hd0 = v + 8 * lhalf;
    size_t base = (size_t)(b * NQ + q_) * D + h * HD;
    O[base + hd0]      = o0[v] * inv;
    O[base + 16 + hd0] = o1[v] * inv;
  }
}

// ---------------------------------------------------------------------------
// Deformable sampling with fused vproj. One wave per (b,h,query); lane = chan.
// ---------------------------------------------------------------------------
__device__ __forceinline__ float bev_at(const float* p, int x, int y) {
  if (x < 0 || x >= WBEV || y < 0 || y >= HBEV) return 0.f;
  return p[y * WBEV + x];
}

__global__ __launch_bounds__(256) void deform_sample(
    const float* __restrict__ bev, const float* __restrict__ off_raw,
    const float* __restrict__ aw_raw, const float* __restrict__ refpos,
    const float* __restrict__ vw, const float* __restrict__ vb,
    float* __restrict__ outp)
{
  int wid  = blockIdx.x * 8 + (threadIdx.x >> 5);
  int lane = threadIdx.x & 31;
  int n = wid & (NQ - 1);
  int h = (wid >> 10) & (NH - 1);
  int b = wid >> 13;
  size_t row = (size_t)b * NQ + n;

  // softmax over P attention weights (uniform across the wave)
  float awv[PP]; float amax = -1e30f;
  #pragma unroll
  for (int p = 0; p < PP; ++p) { awv[p] = aw_raw[row * (NH * PP) + h * PP + p]; amax = fmaxf(amax, awv[p]); }
  float asum = 0.f;
  #pragma unroll
  for (int p = 0; p < PP; ++p) { awv[p] = __expf(awv[p] - amax); asum += awv[p]; }
  float ainv = 1.f / asum;

  float rx = refpos[n * 2 + 0];
  float ry = refpos[n * 2 + 1];
  const float* plane = bev + ((size_t)b * D + h * HD + lane) * (HBEV * WBEV);
  __builtin_prefetch(plane, 0, 1);   // global_prefetch_b8: warm L2 path

  float acc = 0.f;
  #pragma unroll
  for (int p = 0; p < PP; ++p) {
    float ox = off_raw[row * (NH * PP * 2) + (h * PP + p) * 2 + 0];
    float oy = off_raw[row * (NH * PP * 2) + (h * PP + p) * 2 + 1];
    float gx = rx + ox * (1.f / 51.2f);
    float gy = -(ry + oy * (1.f / 51.2f));          // flip y
    float ix = ((gx + 1.f) * WBEV - 1.f) * 0.5f;    // align_corners=False
    float iy = ((gy + 1.f) * HBEV - 1.f) * 0.5f;
    float x0f = floorf(ix), y0f = floorf(iy);
    int x0 = (int)x0f, y0 = (int)y0f;
    float wx1 = ix - x0f, wy1 = iy - y0f;
    float wx0 = 1.f - wx1, wy0 = 1.f - wy1;
    float s = bev_at(plane, x0,     y0    ) * (wx0 * wy0)
            + bev_at(plane, x0 + 1, y0    ) * (wx1 * wy0)
            + bev_at(plane, x0,     y0 + 1) * (wx0 * wy1)
            + bev_at(plane, x0 + 1, y0 + 1) * (wx1 * wy1);
    acc += s * (awv[p] * ainv);
  }

  // Fused vproj: out[o] = vb[o] + sum_c vw[o,c] * acc_c  (ds_bpermute bcast)
  float o = vb[lane];
  #pragma unroll
  for (int c = 0; c < HD; ++c)
    o += vw[lane * HD + c] * __shfl(acc, c, 32);
  outp[row * D + h * HD + lane] = o;
}

// ---------------------------------------------------------------------------
// Init: cur = broadcast(ba_query), embed = sine_embed(ref*GRID), ref output.
// ---------------------------------------------------------------------------
__global__ __launch_bounds__(256) void init_kernel(
    const float* __restrict__ refpos, const float* __restrict__ baq,
    float* __restrict__ cur, float* __restrict__ embed, float* __restrict__ refout)
{
  int idx = blockIdx.x * 256 + threadIdx.x;   // row*256 + d
  int d   = idx & 255;
  int row = idx >> 8;
  int n   = row & (NQ - 1);
  cur[idx] = baq[n * D + d];
  float rpx = refpos[n * 2 + 0] * 51.2f;
  float rpy = refpos[n * 2 + 1] * 51.2f;
  float pos = (d < 128) ? rpy : rpx;          // concat [py, px]
  int j = (d < 128) ? d : d - 128;
  float dim_t = __powf(10000.f, (float)(2 * (j >> 1)) * (1.f / 128.f));
  float ang = pos * 6.283185307179586f / dim_t;
  embed[idx] = (j & 1) ? __cosf(ang) : __sinf(ang);
  if (d < 2) refout[row * 2 + d] = (d == 0) ? rpx : rpy;
}

__global__ __launch_bounds__(256) void add_k(const float* __restrict__ a,
    const float* __restrict__ b, float* __restrict__ o)
{ int i = blockIdx.x * 256 + threadIdx.x; o[i] = a[i] + b[i]; }

__global__ __launch_bounds__(256) void mul_k(const float* __restrict__ a,
    const float* __restrict__ b, float* __restrict__ o)
{ int i = blockIdx.x * 256 + threadIdx.x; o[i] = a[i] * b[i]; }

// LayerNorm(x + y) over last dim (256). One block per row.
__global__ __launch_bounds__(256) void ln_k(const float* __restrict__ x,
    const float* __restrict__ y, const float* __restrict__ g,
    const float* __restrict__ bb, float* __restrict__ o)
{
  __shared__ float red[256];
  int row = blockIdx.x, t = threadIdx.x;
  size_t idx = (size_t)row * D + t;
  float v = x[idx] + y[idx];
  red[t] = v; __syncthreads();
  for (int s = 128; s > 0; s >>= 1) { if (t < s) red[t] += red[t + s]; __syncthreads(); }
  float mean = red[0] * (1.f / D); __syncthreads();
  float dv = v - mean;
  red[t] = dv * dv; __syncthreads();
  for (int s = 128; s > 0; s >>= 1) { if (t < s) red[t] += red[t + s]; __syncthreads(); }
  float var = red[0] * (1.f / D);
  o[idx] = dv * rsqrtf(var + 1e-5f) * g[t] + bb[t];
}

// ---------------------------------------------------------------------------
// Host orchestration.
// Param leaf order = JAX sorted-key pytree flatten of setup_inputs():
//  [0] bev_feat, [1] ba_query,
//  per layer l (base 2+32l): aw{b1,b2,w1,w2} ffn{b1,b2,w1,w2}
//    ln0{b,g} ln1{b,g} ln2{b,g} off{b1,b2,w1,w2} outp{b1,b2,w1,w2}
//    sa{bk,bo,bq,bv,wk,wo,wq,wv} vproj_b vproj_w,
//  [98..101] pos_scale{b1,b2,w1,w2}, [102..105] query_pos{b1,b2,w1,w2}, [106] ref_pos
// ---------------------------------------------------------------------------
extern "C" void kernel_launch(void* const* d_in, const int* in_sizes, int n_in,
                              void* d_out, int out_size, void* d_ws, size_t ws_size,
                              hipStream_t stream)
{
  (void)in_sizes; (void)n_in; (void)out_size; (void)ws_size;
  const float* bev = (const float*)d_in[0];
  const float* baq = (const float*)d_in[1];
  auto LP = [&](int l, int j) { return (const float*)d_in[2 + l * 32 + j]; };
  const float* ps_b1 = (const float*)d_in[98];
  const float* ps_b2 = (const float*)d_in[99];
  const float* ps_w1 = (const float*)d_in[100];
  const float* ps_w2 = (const float*)d_in[101];
  const float* qp_b1 = (const float*)d_in[102];
  const float* qp_b2 = (const float*)d_in[103];
  const float* qp_w1 = (const float*)d_in[104];
  const float* qp_w2 = (const float*)d_in[105];
  const float* refp  = (const float*)d_in[106];

  // Workspace bump allocator (floats).
  char* wsb = (char*)d_ws;
  size_t off = 0;
  auto alloc = [&](size_t nf) {
    float* p = (float*)(wsb + off);
    off += ((nf * sizeof(float) + 255) & ~(size_t)255);
    return p;
  };
  const size_t SZ = (size_t)MROWS * D;          // 2M floats
  float* cur     = alloc(SZ);
  float* posq    = alloc(SZ);
  float* qpos    = alloc(SZ);
  float* tin     = alloc(SZ);                   // also embed / attn out / qd
  float* tmp     = alloc((size_t)MROWS * 512);  // hidden (wide enough for FFN)
  float* qb      = alloc(SZ);                   // also ca_pre
  float* kb      = alloc(SZ);
  float* vbuf    = alloc(SZ);
  float* tgt     = alloc(SZ);
  float* tgt2    = alloc(SZ);
  float* ca      = alloc(SZ);
  float* off_raw = alloc((size_t)MROWS * NH * PP * 2);
  float* aw_raw  = alloc((size_t)MROWS * NH * PP);

  float* out_final = (float*)d_out;
  float* ref_out   = out_final + SZ;

  const int EW_BLOCKS = (int)(SZ / 256);        // elementwise grids

  auto gemm = [&](const float* A, const float* W, const float* bias,
                  float* C, int N, int K, int relu) {
    dim3 g(N / 64, MROWS / 128);
    gemm_bf16<<<g, 256, 0, stream>>>(A, W, bias, C, MROWS, N, K, relu);
  };
  auto eadd = [&](const float* a, const float* b, float* o) {
    add_k<<<EW_BLOCKS, 256, 0, stream>>>(a, b, o);
  };

  // ---- init: cur, embed(tin), ref output; posq = mlp(query_pos, embed) ----
  init_kernel<<<EW_BLOCKS, 256, 0, stream>>>(refp, baq, cur, tin, ref_out);
  gemm(tin, qp_w1, qp_b1, tmp, D, D, 1);
  gemm(tmp, qp_w2, qp_b2, posq, D, D, 0);

  for (int l = 0; l < 3; ++l) {
    // qpos = mlp(pos_scale, cur) * posq
    gemm(cur, ps_w1, ps_b1, tmp, D, D, 1);
    gemm(tmp, ps_w2, ps_b2, qpos, D, D, 0);
    mul_k<<<EW_BLOCKS, 256, 0, stream>>>(qpos, posq, qpos);
    // tgt_in = cur + qpos
    eadd(cur, qpos, tin);
    // self-attention: q,k from tgt_in; v from cur (pre-pos query)
    gemm(tin, LP(l, 28), LP(l, 24), qb,   D, D, 0);   // wq, bq
    gemm(tin, LP(l, 26), LP(l, 22), kb,   D, D, 0);   // wk, bk
    gemm(cur, LP(l, 29), LP(l, 25), vbuf, D, D, 0);   // wv, bv
    attn_flash<<<dim3(NQ / 128, NH, BATCH), 256, 0, stream>>>(qb, kb, vbuf, tin);
    gemm(tin, LP(l, 27), LP(l, 23), tmp, D, D, 0);    // wo, bo -> sa
    ln_k<<<MROWS, 256, 0, stream>>>(cur, tmp, LP(l, 9), LP(l, 8), tgt);   // ln0
    // cross-attn query = tgt + qpos
    eadd(tgt, qpos, tin);
    // offsets + weights MLPs
    gemm(tin, LP(l, 16), LP(l, 14), tmp, D, D, 1);          // off w1 relu
    gemm(tmp, LP(l, 17), LP(l, 15), off_raw, NH * PP * 2, D, 0);
    gemm(tin, LP(l, 2),  LP(l, 0),  tmp, D, D, 1);          // aw w1 relu
    gemm(tmp, LP(l, 3),  LP(l, 1),  aw_raw, NH * PP, D, 0);
    // fused vproj + bilinear sample + weighted sum -> ca_pre (qb)
    deform_sample<<<MROWS * NH / 8, 256, 0, stream>>>(
        bev, off_raw, aw_raw, refp, LP(l, 31), LP(l, 30), qb);
    // output-projection MLP
    gemm(qb,  LP(l, 20), LP(l, 18), tmp, D, D, 1);
    gemm(tmp, LP(l, 21), LP(l, 19), ca,  D, D, 0);
    ln_k<<<MROWS, 256, 0, stream>>>(tgt, ca, LP(l, 11), LP(l, 10), tgt2);  // ln1
    // FFN
    gemm(tgt2, LP(l, 6), LP(l, 4), tmp, 512, D, 1);
    gemm(tmp,  LP(l, 7), LP(l, 5), ca,  D, 512, 0);
    float* dst = (l == 2) ? out_final : cur;
    ln_k<<<MROWS, 256, 0, stream>>>(tgt2, ca, LP(l, 13), LP(l, 12), dst);  // ln2
  }
}
